// PointResidualEncoder_52561809768829
// MI455X (gfx1250) — matrix operand order repaired and checked by reading, not compiled, not run on hardware
//
#include <hip/hip_runtime.h>
#include <hip/hip_bf16.h>

typedef __attribute__((ext_vector_type(2)))  float    v2f;
typedef __attribute__((ext_vector_type(8)))  float    v8f;
typedef __attribute__((ext_vector_type(16))) _Float16 v16h;

#define N_ORIG      262144
#define M_COARSE    65536
#define KNN         16
#define THRES       8.0f
#define CAND_TILES  (N_ORIG / 16)    // 16384
#define QUERY_TILES (M_COARSE / 16)  // 4096
#define CH          32               // candidate tiles staged per chunk (8 KB)

// ---------------------------------------------------------------------------
// Prep: candidate A-tiles for V_WMMA_F32_16X16X4_F32.
// A row m = (-2x, -2y, -2z, |x|^2)  so  A·[qx,qy,qz,1]^T = |x|^2 - 2 q·x.
// f32 A 16x4 layout: VGPR0 lanes0-15=K0, lanes16-31=K2; VGPR1 K1/K3.
// Store per-lane float2 so one b64 load fills both VGPRs.
// ---------------------------------------------------------------------------
__global__ void prep_cand(const float* __restrict__ xo, float2* __restrict__ candA) {
    int t = blockIdx.x * blockDim.x + threadIdx.x;
    if (t >= CAND_TILES * 32) return;
    int tile = t >> 5, lane = t & 31;
    int m = tile * 16 + (lane & 15);
    float x = xo[m * 3 + 0], y = xo[m * 3 + 1], z = xo[m * 3 + 2];
    float2 v = (lane < 16) ? make_float2(-2.f * x, -2.f * y)
                           : make_float2(-2.f * z, x * x + y * y + z * z);
    candA[tile * 32 + lane] = v;
}

// Query B-tiles (4x16 f32 B): per-lane float2 = (K0,K1) for lanes<16 = (qx,qy),
// (K2,K3) for lanes>=16 = (qz, 1).
__global__ void prep_query(const float* __restrict__ xc, float2* __restrict__ queryB) {
    int t = blockIdx.x * blockDim.x + threadIdx.x;
    if (t >= QUERY_TILES * 32) return;
    int tile = t >> 5, lane = t & 31;
    int q = tile * 16 + (lane & 15);
    float x = xc[q * 3 + 0], y = xc[q * 3 + 1], z = xc[q * 3 + 2];
    float2 v = (lane < 16) ? make_float2(x, y) : make_float2(z, 1.0f);
    queryB[tile * 32 + lane] = v;
}

// Weight B-tiles.
//  w1B: f32 4x16 B tiles of [W1; b1] (bias folded via the residual's 4th coord = 1).
//  w2B/w3B: f16 32x16 B tiles, lane l<16: column n=ct*16+l, K=kt*32+0..15;
//           lane>=16: same column, K=kt*32+16..31. 16 f16 contiguous per lane.
__global__ void prep_weights(const float* __restrict__ W1, const float* __restrict__ b1,
                             const float* __restrict__ W2, const float* __restrict__ W3,
                             float2* __restrict__ w1B,
                             _Float16* __restrict__ w2B, _Float16* __restrict__ w3B) {
    int tid = blockIdx.x * blockDim.x + threadIdx.x;
    if (tid < 4 * 32) {                               // W1 aug, f32 tiles
        int ct = tid >> 5, lane = tid & 31;
        int c = ct * 16 + (lane & 15);
        float2 v = (lane < 16) ? make_float2(W1[0 * 64 + c], W1[1 * 64 + c])
                               : make_float2(W1[2 * 64 + c], b1[c]);
        w1B[ct * 32 + lane] = v;
    } else if (tid < 128 + 16 * 32) {                 // W2 [64,128] -> 2x8 tiles
        int t = tid - 128;
        int tile = t >> 5, lane = t & 31;
        int kt = tile >> 3, ct = tile & 7;
        int n = ct * 16 + (lane & 15);
        int kbase = kt * 32 + ((lane < 16) ? 0 : 16);
        #pragma unroll
        for (int r = 0; r < 16; ++r)
            w2B[(tile * 32 + lane) * 16 + r] = (_Float16)W2[(kbase + r) * 128 + n];
    } else if (tid < 128 + 512 + 64 * 32) {           // W3 [128,256] -> 4x16 tiles
        int t = tid - 640;
        int tile = t >> 5, lane = t & 31;
        int kt = tile >> 4, ct = tile & 15;
        int n = ct * 16 + (lane & 15);
        int kbase = kt * 32 + ((lane < 16) ? 0 : 16);
        #pragma unroll
        for (int r = 0; r < 16; ++r)
            w3B[(tile * 32 + lane) * 16 + r] = (_Float16)W3[(kbase + r) * 256 + n];
    }
}

// ---------------------------------------------------------------------------
// kNN: one wave per 16-query tile, 8 waves per block. All 8 waves scan the
// SAME candidate stream, so the block stages 32-tile chunks of candidate
// A-tiles into LDS (double-buffered: next chunk loads to registers while the
// current chunk is consumed). The inner loop is software-pipelined: the next
// tile's ds_load_b64 is issued before the current tile's WMMA/compares so the
// wait at the WMMA is dscnt<=1, not a full LDS round-trip. Lane q and lane
// q+16 keep sorted top-16 over disjoint candidate rows (C-layout), merged
// through LDS at the end. Then the +/-8 outlier fallback is applied and
// residual A-tiles are emitted (f32 16x4 layout, 4th coord = 1 folds b1).
// ---------------------------------------------------------------------------
__global__ __launch_bounds__(256) void knn_kernel(
    const float* __restrict__ xo, const float* __restrict__ xc,
    const float2* __restrict__ candA, const float2* __restrict__ queryB,
    float2* __restrict__ resA) {
    __shared__ float2 stage[2][CH * 32];   // 2 x 8 KB candidate chunks
    __shared__ float  sd[8][32][KNN];
    __shared__ int    si[8][32][KNN];
    __shared__ int    mi[8][16][KNN];

    int tid  = threadIdx.x;
    int w    = tid >> 5;
    int lane = tid & 31;
    int qt   = blockIdx.x * 8 + w;

    float2 bq = queryB[qt * 32 + lane];
    v2f b; b[0] = bq.x; b[1] = bq.y;

    float bd[KNN]; int bi[KNN];
    #pragma unroll
    for (int s = 0; s < KNN; ++s) { bd[s] = 3.0e38f; bi[s] = 0; }

    int rbase = (lane < 16) ? 0 : 8;
    const int nchunks = CAND_TILES / CH;               // 512

    // cooperative staging: 256 threads x 2 float4 = one 8 KB chunk
    const float4* gsrc = (const float4*)candA;
    float4*       sbuf0 = (float4*)&stage[0][0];
    float4*       sbuf1 = (float4*)&stage[1][0];

    float4 r0 = gsrc[tid * 2 + 0];
    float4 r1 = gsrc[tid * 2 + 1];
    sbuf0[tid * 2 + 0] = r0;
    sbuf0[tid * 2 + 1] = r1;
    __syncthreads();

    int buf = 0;
    for (int c = 0; c < nchunks; ++c) {
        if (c + 1 < nchunks) {                         // issue next chunk's loads
            int gb = (c + 1) * CH * 16;                // chunk base in float4 units
            r0 = gsrc[gb + tid * 2 + 0];
            r1 = gsrc[gb + tid * 2 + 1];
        }
        const float2* st = &stage[buf][0];
        int base = c * CH * 16 + rbase;
        float2 af = st[lane];                          // tile 0 of this chunk
        #pragma unroll 4
        for (int tt = 0; tt < CH; ++tt) {
            // issue next tile's ds_load before consuming the current one
            float2 nf = st[((tt + 1) & (CH - 1)) * 32 + lane];
            v2f a; a[0] = af.x; a[1] = af.y;
            v8f cacc = {};
            cacc = __builtin_amdgcn_wmma_f32_16x16x4_f32(false, a, false, b,
                                                         (short)0, cacc, false, false);
            #pragma unroll
            for (int j = 0; j < 8; ++j) {
                float d = cacc[j];
                if (d < bd[KNN - 1]) {                 // rare: single bubble insert
                    bd[KNN - 1] = d; bi[KNN - 1] = base + tt * 16 + j;
                    #pragma unroll
                    for (int s = KNN - 1; s > 0; --s) {
                        float dp = bd[s - 1]; int ip = bi[s - 1];
                        bool sw = bd[s] < dp;
                        bd[s - 1] = sw ? bd[s] : dp;    bi[s - 1] = sw ? bi[s] : ip;
                        bd[s]     = sw ? dp    : bd[s]; bi[s]     = sw ? ip    : bi[s];
                    }
                }
            }
            af = nf;
        }
        __syncthreads();
        if (c + 1 < nchunks) {                         // publish next chunk
            float4* dbuf = buf ? sbuf0 : sbuf1;
            dbuf[tid * 2 + 0] = r0;
            dbuf[tid * 2 + 1] = r1;
        }
        __syncthreads();
        buf ^= 1;
    }

    #pragma unroll
    for (int s = 0; s < KNN; ++s) { sd[w][lane][s] = bd[s]; si[w][lane][s] = bi[s]; }
    __syncthreads();

    if (lane < 16) {
        int ia = 0, ib = 0;
        for (int k = 0; k < KNN; ++k) {                // merge two sorted half-lists
            float da = sd[w][lane][ia], db = sd[w][lane + 16][ib];
            bool ta = da <= db;
            mi[w][lane][k] = ta ? si[w][lane][ia] : si[w][lane + 16][ib];
            ia += ta ? 1 : 0; ib += ta ? 0 : 1;
        }
        int q = qt * 16 + lane;
        float qx = xc[q * 3 + 0], qy = xc[q * 3 + 1], qz = xc[q * 3 + 2];
        int i0 = mi[w][lane][0];
        float r0x = xo[i0 * 3 + 0] - qx, r0y = xo[i0 * 3 + 1] - qy, r0z = xo[i0 * 3 + 2] - qz;
        for (int j = 0; j < KNN; ++j) {
            int id = mi[w][lane][j];
            float rx = xo[id * 3 + 0] - qx, ry = xo[id * 3 + 1] - qy, rz = xo[id * 3 + 2] - qz;
            if (fabsf(rx) > THRES || fabsf(ry) > THRES || fabsf(rz) > THRES) {
                rx = r0x; ry = r0y; rz = r0z;          // fallback to nearest neighbor
            }
            resA[q * 32 + j]      = make_float2(rx, ry);
            resA[q * 32 + 16 + j] = make_float2(rz, 1.0f);
        }
    }
}

// ---------------------------------------------------------------------------
// MLP + max-pool: one wave handles one coarse point's 16x3 residual tile.
// L1: f32 WMMA 16x16x4 (bias folded).  L2/L3: f16 WMMA 16x16x32, f32 accum.
// Activations re-swizzled C-layout -> A-layout through per-wave LDS.
// Max over K = column max of the 16x256 tile (7 maxes + shfl_xor 16).
// ---------------------------------------------------------------------------
__global__ __launch_bounds__(256) void mlp_kernel(
    const float2* __restrict__ resA, const float2* __restrict__ w1B,
    const _Float16* __restrict__ w2B, const _Float16* __restrict__ w3B,
    const float* __restrict__ b2, const float* __restrict__ b3,
    float* __restrict__ out) {
    __shared__ _Float16 hbuf[8][16 * 128];
    int w = threadIdx.x >> 5, lane = threadIdx.x & 31;
    _Float16* hw = hbuf[w];
    int wave = blockIdx.x * 8 + w;
    int col = lane & 15;                 // also A-row for this lane
    int rbase = (lane < 16) ? 0 : 8;
    int kadd = (lane < 16) ? 0 : 16;

    for (int it = 0; it < 4; ++it) {
        int p = wave + it * 16384;
        float2 av = resA[p * 32 + lane];
        v2f a1; a1[0] = av.x; a1[1] = av.y;

        // ---- layer 1: [16x4] x [4x64] f32 WMMA, relu, f16 -> LDS (ld=128)
        #pragma unroll
        for (int ct = 0; ct < 4; ++ct) {
            float2 wv = w1B[ct * 32 + lane];
            v2f bw; bw[0] = wv.x; bw[1] = wv.y;
            v8f c = {};
            c = __builtin_amdgcn_wmma_f32_16x16x4_f32(false, a1, false, bw,
                                                      (short)0, c, false, false);
            #pragma unroll
            for (int j = 0; j < 8; ++j)
                hw[(rbase + j) * 128 + ct * 16 + col] = (_Float16)fmaxf(c[j], 0.0f);
        }

        // ---- layer 2: [16x64] x [64x128], 2 K-tiles x 8 col-tiles
        v16h a2[2];
        #pragma unroll
        for (int kt = 0; kt < 2; ++kt) {
            const _Float16* s = hw + col * 128 + kt * 32 + kadd;
            v16h a;
            ((float4*)&a)[0] = *(const float4*)(s);        // K 0..7   (or 16..23)
            ((float4*)&a)[1] = *(const float4*)(s + 8);    // K 8..15  (or 24..31)
            a2[kt] = a;
        }
        #pragma unroll
        for (int ct = 0; ct < 8; ++ct) {
            v8f c = {};
            #pragma unroll
            for (int kt = 0; kt < 2; ++kt) {
                v16h bw = *(const v16h*)(w2B + ((kt * 8 + ct) * 32 + lane) * 16);
                c = __builtin_amdgcn_wmma_f32_16x16x32_f16(false, a2[kt], false, bw,
                                                           (short)0, c, false, false);
            }
            float bb = b2[ct * 16 + col];
            #pragma unroll
            for (int j = 0; j < 8; ++j)
                hw[(rbase + j) * 128 + ct * 16 + col] = (_Float16)fmaxf(c[j] + bb, 0.0f);
        }

        // ---- layer 3: [16x128] x [128x256], 4 K-tiles x 16 col-tiles + col-max
        v16h a3[4];
        #pragma unroll
        for (int kt = 0; kt < 4; ++kt) {
            const _Float16* s = hw + col * 128 + kt * 32 + kadd;
            v16h a;
            ((float4*)&a)[0] = *(const float4*)(s);
            ((float4*)&a)[1] = *(const float4*)(s + 8);
            a3[kt] = a;
        }
        #pragma unroll
        for (int ct = 0; ct < 16; ++ct) {
            v8f c = {};
            #pragma unroll
            for (int kt = 0; kt < 4; ++kt) {
                v16h bw = *(const v16h*)(w3B + ((kt * 16 + ct) * 32 + lane) * 16);
                c = __builtin_amdgcn_wmma_f32_16x16x32_f16(false, a3[kt], false, bw,
                                                           (short)0, c, false, false);
            }
            float m = c[0];
            #pragma unroll
            for (int j = 1; j < 8; ++j) m = fmaxf(m, c[j]);   // rows rbase..rbase+7
            m = fmaxf(m, __shfl_xor(m, 16, 32));              // combine both halves
            if (lane < 16) out[p * 256 + ct * 16 + lane] = m + b3[ct * 16 + lane];
        }
    }
}

// ---------------------------------------------------------------------------
extern "C" void kernel_launch(void* const* d_in, const int* in_sizes, int n_in,
                              void* d_out, int out_size, void* d_ws, size_t ws_size,
                              hipStream_t stream) {
    (void)in_sizes; (void)n_in; (void)out_size; (void)ws_size;
    const float* xo = (const float*)d_in[0];
    const float* xc = (const float*)d_in[1];
    const float* W1 = (const float*)d_in[2];
    const float* b1 = (const float*)d_in[3];
    const float* W2 = (const float*)d_in[4];
    const float* b2 = (const float*)d_in[5];
    const float* W3 = (const float*)d_in[6];
    const float* b3 = (const float*)d_in[7];

    char* ws = (char*)d_ws;
    size_t off = 0;
    auto carve = [&](size_t bytes) {
        void* p = ws + off;
        off = (off + bytes + 255) & ~(size_t)255;
        return p;
    };
    float2*    candA  = (float2*)   carve((size_t)CAND_TILES  * 32 * sizeof(float2)); // 4 MB
    float2*    queryB = (float2*)   carve((size_t)QUERY_TILES * 32 * sizeof(float2)); // 1 MB
    float2*    resA   = (float2*)   carve((size_t)M_COARSE    * 32 * sizeof(float2)); // 16 MB
    float2*    w1B    = (float2*)   carve(4  * 32 * sizeof(float2));
    _Float16*  w2B    = (_Float16*) carve(16 * 32 * 16 * sizeof(_Float16));
    _Float16*  w3B    = (_Float16*) carve(64 * 32 * 16 * sizeof(_Float16));

    prep_cand   <<<CAND_TILES * 32 / 256, 256, 0, stream>>>(xo, candA);
    prep_query  <<<QUERY_TILES * 32 / 256, 256, 0, stream>>>(xc, queryB);
    prep_weights<<<11, 256, 0, stream>>>(W1, b1, W2, W3, w1B, w2B, w3B);
    knn_kernel  <<<QUERY_TILES / 8, 256, 0, stream>>>(xo, xc, candA, queryB, resA);
    mlp_kernel  <<<2048, 256, 0, stream>>>(resA, w1B, w2B, w3B, b2, b3, (float*)d_out);
}
